// WindowedGlobalAttention3D_24678882082930
// MI455X (gfx1250) — matrix operand order repaired and checked by reading, not compiled, not run on hardware
//
#include <hip/hip_runtime.h>
#include <hip/hip_bf16.h>

typedef _Float16 half_t;
typedef __attribute__((ext_vector_type(16))) _Float16 v16h;
typedef __attribute__((ext_vector_type(8)))  _Float16 v8h;
typedef __attribute__((ext_vector_type(8)))  float    v8f;

// ---------------------------------------------------------------------------
// WMMA fragment helpers (16x16x32 f16; contiguous k order used by this code for A and B)
// A 16x32 (MxK): lane m=l%16; lanes 0-15 hold K k0..k0+15 (contiguous),
//                lanes 16-31 hold K k0+16..k0+31.
// B 32x16 (KxN): lane n=l%16; lanes 0-15 hold K k0..k0+15 (contiguous),
//                lanes 16-31 hold K k0+16..k0+31.
// C/D 16x16 f32: lane n=l%16; VGPR r holds row m = 8*(l/16) + r.
// ---------------------------------------------------------------------------
__device__ __forceinline__ v16h wma_load_a(const half_t* buf, int lda, int k0, int lane) {
  int m   = lane & 15;
  int klo = k0 + ((lane >> 4) << 4);
  const half_t* p = buf + m * lda + klo;
  union { v16h v; v8h q[2]; } u;
  u.q[0] = *(const v8h*)(p);
  u.q[1] = *(const v8h*)(p + 8);
  return u.v;
}

__device__ __forceinline__ v16h wma_load_b(const half_t* buf, int ldb, int n0, int k0, int lane) {
  int n  = n0 + (lane & 15);
  int kb = k0 + ((lane >> 4) << 4);
  return *(const v16h*)(buf + n * ldb + kb);
}

__device__ __forceinline__ v8f wmma16(v16h a, v16h b, v8f c) {
  return __builtin_amdgcn_wmma_f32_16x16x32_f16(false, a, false, b, (short)0, c, false, false);
}

// ---------------------------------------------------------------------------
// Prep: transpose f32 [K][N] weight -> f16 [N][K] so WMMA B-fragments are
// single contiguous 32B loads per lane.
// ---------------------------------------------------------------------------
__global__ void wga3d_prep_transpose(const float* __restrict__ src,
                                     half_t* __restrict__ dst, int K, int N) {
  int t = blockIdx.x * blockDim.x + threadIdx.x;
  if (t >= K * N) return;
  int k = t / N, n = t % N;
  dst[n * K + k] = (half_t)src[t];
}

// ---------------------------------------------------------------------------
// Fused windowed attention: one wave32 per 16-token window.
// Dims hardcoded: B=2, C=64, D=32, H=W=64, WS=4, S=16, ED=128, NH=2, HD=64, OD=64.
// ---------------------------------------------------------------------------
#define WAVES_PER_BLOCK 4
#define LDS_PER_WAVE    28672

__global__ void wga3d_main(const float* __restrict__ x,
                           const float* __restrict__ pos,
                           const half_t* __restrict__ WcT,   // [128][64]
                           const float* __restrict__ bc,
                           const half_t* __restrict__ WqT,   // [128][128]
                           const half_t* __restrict__ WkT,
                           const half_t* __restrict__ WvT,
                           const half_t* __restrict__ WoT,
                           const float* __restrict__ bo,
                           const float* __restrict__ ln_g,
                           const float* __restrict__ ln_b,
                           const half_t* __restrict__ WoutT, // [64][128]
                           const float* __restrict__ bout,
                           float* __restrict__ out) {
  extern __shared__ char smem[];
  const int lane = threadIdx.x & 31;
  const int wave = threadIdx.x >> 5;
  char* base = smem + wave * LDS_PER_WAVE;

  float*  xp     = (float*)(base);            // 16x128 f32: residual, later post-Wo
  half_t* xph    = (half_t*)(base + 8192);    // 16x128 f16: xp+pos, later LN output
  half_t* q_h    = (half_t*)(base + 12288);   // 16x128 f16: Q, later attn@V output
  half_t* k_h    = (half_t*)(base + 16384);   // 16x128 f16
  half_t* v_t    = (half_t*)(base + 20480);   // 128x16 f16 (V transposed)
  half_t* xin    = (half_t*)(base + 24576);   // 16x64 f16 (window gather)
  half_t* attn_h = (half_t*)(base + 24576);   // 2x16x16 f16 (reuses xin)
  float*  attn_f = (float*)(base + 26624);    // 2x16x16 f32

  const int widx = blockIdx.x * WAVES_PER_BLOCK + wave;   // 0..16383
  const int b    = widx >> 13;          // / 8192 windows per batch
  const int rem  = widx & 8191;
  const int dpt  = rem >> 8;            // depth slice
  const int r2   = rem & 255;
  const int hy   = r2 >> 4;
  const int wxw  = r2 & 15;
  const int h0   = hy * 4, w0 = wxw * 4;

  // ---- gather window (B,C,D,H,W) -> xin[16 seq][64 ch] f16 -----------------
  const float* xb = x + (size_t)b * 64 * 32 * 64 * 64;
  for (int it = 0; it < 32; ++it) {
    int c  = it * 2 + (lane >> 4);
    int s  = lane & 15;
    int hh = h0 + (s >> 2);
    int ww = w0 + (s & 3);
    xin[s * 64 + c] = (half_t)xb[(((size_t)c * 32 + dpt) * 64 + hh) * 64 + ww];
  }

  // ---- GEMM1: xp = win @ Wc + bc ; xph = f16(xp + pos) --------------------
  for (int nt = 0; nt < 8; ++nt) {
    v8f acc = {};
    for (int kt = 0; kt < 2; ++kt)
      acc = wmma16(wma_load_a(xin, 64, kt * 32, lane),
                   wma_load_b(WcT, 64, nt * 16, kt * 32, lane), acc);
    int n = nt * 16 + (lane & 15);
    int mb = (lane >> 4) << 3;
    float bcn = bc[n];
#pragma unroll
    for (int r = 0; r < 8; ++r) {
      int m = mb + r;
      float xv = acc[r] + bcn;
      xp[m * 128 + n]  = xv;                         // residual (pre-pos)
      xph[m * 128 + n] = (half_t)(xv + pos[m * 128 + n]);
    }
  }

  // ---- GEMM2/3/4: Q (pre-scaled by 1/sqrt(HD)=0.125), K, V^T --------------
  for (int nt = 0; nt < 8; ++nt) {
    v8f aq = {}, ak = {}, av = {};
    for (int kt = 0; kt < 4; ++kt) {
      v16h a = wma_load_a(xph, 128, kt * 32, lane);
      aq = wmma16(a, wma_load_b(WqT, 128, nt * 16, kt * 32, lane), aq);
      ak = wmma16(a, wma_load_b(WkT, 128, nt * 16, kt * 32, lane), ak);
      av = wmma16(a, wma_load_b(WvT, 128, nt * 16, kt * 32, lane), av);
    }
    int n = nt * 16 + (lane & 15);
    int mb = (lane >> 4) << 3;
#pragma unroll
    for (int r = 0; r < 8; ++r) {
      int m = mb + r;
      q_h[m * 128 + n] = (half_t)(aq[r] * 0.125f);
      k_h[m * 128 + n] = (half_t)ak[r];
      v_t[n * 16 + m]  = (half_t)av[r];              // transposed for attn@V B-frags
    }
  }

  // ---- logits: per head, Q(16x64) @ K^T -> attn_f[2][16][16] --------------
  for (int hd = 0; hd < 2; ++hd) {
    v8f acc = {};
    for (int kt = 0; kt < 2; ++kt)
      acc = wmma16(wma_load_a(q_h, 128, hd * 64 + kt * 32, lane),
                   wma_load_b(k_h, 128, 0, hd * 64 + kt * 32, lane), acc);
    int n = lane & 15;
    int mb = (lane >> 4) << 3;
#pragma unroll
    for (int r = 0; r < 8; ++r)
      attn_f[(hd * 16 + mb + r) * 16 + n] = acc[r];
  }

  // ---- softmax: lane = head*16 + query row --------------------------------
  {
    const float* row = attn_f + lane * 16;
    float mx = row[0];
#pragma unroll
    for (int j = 1; j < 16; ++j) mx = fmaxf(mx, row[j]);
    float e[16], sum = 0.f;
#pragma unroll
    for (int j = 0; j < 16; ++j) { e[j] = expf(row[j] - mx); sum += e[j]; }
    float inv = 1.f / sum;
#pragma unroll
    for (int j = 0; j < 16; ++j) attn_h[lane * 16 + j] = (half_t)(e[j] * inv);
  }

  // ---- attn @ V (K=16, zero-padded to 32); output overwrites q_h ----------
  for (int hd = 0; hd < 2; ++hd) {
    union { v16h v; v8h q[2]; } ua;
    {
      int m = lane & 15;
      if (lane < 16) {
        ua.v = *(const v16h*)(attn_h + (hd * 16 + m) * 16);    // K=0..15
      } else {
#pragma unroll
        for (int i = 0; i < 16; ++i) ua.v[i] = (half_t)0.f;    // K=16..31 pad
      }
    }
    for (int t = 0; t < 4; ++t) {
      v16h bfr;
      int ch = hd * 64 + t * 16 + (lane & 15);
      if (lane < 16) {
        bfr = *(const v16h*)(v_t + ch * 16);                   // K=0..15
      } else {
#pragma unroll
        for (int i = 0; i < 16; ++i) bfr[i] = (half_t)0.f;     // K=16..31 pad
      }
      v8f acc = {};
      acc = wmma16(ua.v, bfr, acc);
      int n = hd * 64 + t * 16 + (lane & 15);
      int mb = (lane >> 4) << 3;
#pragma unroll
      for (int r = 0; r < 8; ++r)
        q_h[(mb + r) * 128 + n] = (half_t)acc[r];
    }
  }

  // ---- GEMM7: o @ Wo + bo + residual (in place into xp) -------------------
  for (int nt = 0; nt < 8; ++nt) {
    v8f acc = {};
    for (int kt = 0; kt < 4; ++kt)
      acc = wmma16(wma_load_a(q_h, 128, kt * 32, lane),
                   wma_load_b(WoT, 128, nt * 16, kt * 32, lane), acc);
    int n = nt * 16 + (lane & 15);
    int mb = (lane >> 4) << 3;
    float bon = bo[n];
#pragma unroll
    for (int r = 0; r < 8; ++r)
      xp[(mb + r) * 128 + n] += acc[r] + bon;
  }

  // ---- LayerNorm over 128: lane = (row m, half) ; combine via shfl_xor ----
  {
    int m  = lane & 15;
    int hf = lane >> 4;
    const float* row = xp + m * 128 + hf * 64;
    float s = 0.f;
    for (int j = 0; j < 64; ++j) s += row[j];
    s += __shfl_xor(s, 16, 32);
    float mu = s * (1.f / 128.f);
    float v2 = 0.f;
    for (int j = 0; j < 64; ++j) { float t = row[j] - mu; v2 += t * t; }
    v2 += __shfl_xor(v2, 16, 32);
    float rs = rsqrtf(v2 * (1.f / 128.f) + 1e-5f);
    for (int j = 0; j < 64; ++j) {
      int n = hf * 64 + j;
      xph[m * 128 + n] = (half_t)((row[j] - mu) * rs * ln_g[n] + ln_b[n]);
    }
  }

  // ---- GEMM8: ln @ Wout + bout, scatter with reference's exact view -------
  // flat (s,oc) of the 16x64 tile is re-viewed as (OD=64, 4, 4):
  //   out_channel = s*4 + oc/16 ; dh = (oc>>2)&3 ; dw = oc&3
  float* ob = out + (size_t)b * 64 * 32 * 64 * 64;
  for (int nt = 0; nt < 4; ++nt) {
    v8f acc = {};
    for (int kt = 0; kt < 4; ++kt)
      acc = wmma16(wma_load_a(xph, 128, kt * 32, lane),
                   wma_load_b(WoutT, 128, nt * 16, kt * 32, lane), acc);
    int oc = nt * 16 + (lane & 15);
    int mb = (lane >> 4) << 3;
    float bn = bout[oc];
#pragma unroll
    for (int r = 0; r < 8; ++r) {
      int s  = mb + r;
      int o  = s * 4 + (oc >> 4);
      int hh = h0 + ((oc >> 2) & 3);
      int ww = w0 + (oc & 3);
      ob[(((size_t)o * 32 + dpt) * 64 + hh) * 64 + ww] = acc[r] + bn;
    }
  }
}

// ---------------------------------------------------------------------------
extern "C" void kernel_launch(void* const* d_in, const int* in_sizes, int n_in,
                              void* d_out, int out_size, void* d_ws, size_t ws_size,
                              hipStream_t stream) {
  const float* x    = (const float*)d_in[0];
  const float* pos  = (const float*)d_in[1];
  const float* Wc   = (const float*)d_in[2];
  const float* bc   = (const float*)d_in[3];
  const float* Wq   = (const float*)d_in[4];
  const float* Wk   = (const float*)d_in[5];
  const float* Wv   = (const float*)d_in[6];
  const float* Wo   = (const float*)d_in[7];
  const float* bo   = (const float*)d_in[8];
  const float* lng  = (const float*)d_in[9];
  const float* lnb  = (const float*)d_in[10];
  const float* Wout = (const float*)d_in[11];
  const float* bout = (const float*)d_in[12];

  half_t* ws     = (half_t*)d_ws;
  half_t* WcT    = ws;                  // 128*64
  half_t* WqT    = WcT  + 128 * 64;     // 128*128
  half_t* WkT    = WqT  + 128 * 128;
  half_t* WvT    = WkT  + 128 * 128;
  half_t* WoT    = WvT  + 128 * 128;
  half_t* WoutT  = WoT  + 128 * 128;    // 64*128

  auto T = [&](const float* src, half_t* dst, int K, int N) {
    int total = K * N;
    wga3d_prep_transpose<<<(total + 255) / 256, 256, 0, stream>>>(src, dst, K, N);
  };
  T(Wc,   WcT,   64, 128);
  T(Wq,   WqT,  128, 128);
  T(Wk,   WkT,  128, 128);
  T(Wv,   WvT,  128, 128);
  T(Wo,   WoT,  128, 128);
  T(Wout, WoutT, 128, 64);

  const int total_windows = 2 * 32 * 16 * 16;       // 16384
  const int blocks = total_windows / WAVES_PER_BLOCK;
  const size_t lds = (size_t)WAVES_PER_BLOCK * LDS_PER_WAVE;   // 112 KB dynamic LDS
  wga3d_main<<<blocks, WAVES_PER_BLOCK * 32, lds, stream>>>(
      x, pos, WcT, bc, WqT, WkT, WvT, WoT, bo, lng, lnb, WoutT, bout,
      (float*)d_out);
}
